// GraphAttentionLayer_EFA_15195594293710
// MI455X (gfx1250) — compile-verified, matrix operand order checked by reading, not run on hardware
//
#include <hip/hip_runtime.h>
#include <math.h>

typedef _Float16 v16h __attribute__((ext_vector_type(16)));
typedef _Float16 v8h  __attribute__((ext_vector_type(8)));
typedef float    v8f  __attribute__((ext_vector_type(8)));

#define ALPHA   0.2f
#define NEG_INF -9e15f

// lrelu(x) = max(x, 0.2x)  (packable: v_pk_mul + v_pk_max)
__device__ __forceinline__ float lrelu(float x) { return fmaxf(x, ALPHA * x); }

__device__ __forceinline__ v16h combine8(v8h lo, v8h hi) {
    return __builtin_shufflevector(lo, hi, 0,1,2,3,4,5,6,7,8,9,10,11,12,13,14,15);
}

// ---------------------------------------------------------------------------
// Kernel 1: fold ah_w into wh1_w / wh2_w  ->  v1[128], v2[128], b12[2]
// ---------------------------------------------------------------------------
__global__ void fold_kernel(const float* __restrict__ wh1_w, const float* __restrict__ wh1_b,
                            const float* __restrict__ wh2_w, const float* __restrict__ wh2_b,
                            const float* __restrict__ ah_w,  const float* __restrict__ ah_b,
                            float* __restrict__ v1, float* __restrict__ v2,
                            float* __restrict__ b12) {
    int k = threadIdx.x;                       // 0..127
    float s1 = 0.f, s2 = 0.f;
    for (int o = 0; o < 64; ++o) {
        float a = ah_w[o];
        s1 += wh1_w[k * 64 + o] * a;
        s2 += wh2_w[k * 64 + o] * a;
    }
    v1[k] = s1; v2[k] = s2;
    if (k == 0) {
        float t1 = 0.f, t2 = 0.f;
        for (int o = 0; o < 64; ++o) { t1 += wh1_b[o] * ah_w[o]; t2 += wh2_b[o] * ah_w[o]; }
        b12[0] = t1 + ah_b[0];                 // row term carries ah_b once
        b12[1] = t2;
    }
}

// ---------------------------------------------------------------------------
// Kernel 2: per (b,n): hT16[b][o][n] = f16(input@wh_w + wh_b)  (transposed f16
// copy of h so the attention GEMM's B-fragments load contiguously),
// e1 = input.v1 + b1, e2 = input.v2 + b2
// ---------------------------------------------------------------------------
__global__ __launch_bounds__(128)
void proj_kernel(const float* __restrict__ input, const float* __restrict__ wh_w,
                 const float* __restrict__ wh_b,  const float* __restrict__ v1,
                 const float* __restrict__ v2,    const float* __restrict__ b12,
                 _Float16* __restrict__ hT16, float* __restrict__ e1, float* __restrict__ e2) {
    __shared__ float row[128];
    const int bn = blockIdx.x;                 // b*1024 + n
    const int t  = threadIdx.x;
    row[t] = input[(size_t)bn * 128 + t];
    __syncthreads();
    if (t < 64) {
        float acc = wh_b[t];
        for (int k = 0; k < 128; ++k) acc += row[k] * wh_w[k * 64 + t];
        const int b = bn >> 10, n = bn & 1023;
        hT16[(size_t)b * 65536 + (size_t)t * 1024 + n] = (_Float16)acc;
    } else if (t == 64) {
        float a = b12[0];
        for (int k = 0; k < 128; ++k) a += row[k] * v1[k];
        e1[bn] = a;
    } else if (t == 65) {
        float a = b12[1];
        for (int k = 0; k < 128; ++k) a += row[k] * v2[k];
        e2[bn] = a;
    }
}

// ---------------------------------------------------------------------------
// Kernel 3 (heavy, WMMA): per (b,i) row, stream edge_feat[b,i,:,:]:
//   scores[b,i,j] = adj ? lrelu(e1+e2) + lrelu(lrelu(ef@wf_w + wf_b + bf)@af_w + af_b)
//                       : NEG_INF
// 8 waves x 8 j-tiles of 16 edges. Per tile: A = 16x32 f16 (K=16 zero-padded),
// 4x v_wmma_f32_16x16x32_f16 with bias preloaded into C, lrelu+af_w applied
// per lane, 16-lane butterfly o-reduction, then 16 lanes emit 16 coalesced
// scores (cndmask row select).
// ---------------------------------------------------------------------------
__global__ __launch_bounds__(256)
void score_kernel(const float* __restrict__ edge_feat, const int* __restrict__ adj,
                  const float* __restrict__ wf_w, const float* __restrict__ wf_b,
                  const float* __restrict__ af_w, const float* __restrict__ af_b_,
                  const float* __restrict__ bf,   const float* __restrict__ e1_,
                  const float* __restrict__ e2_,  float* __restrict__ scores) {
    const int bi    = blockIdx.x;              // b*1024 + i
    const int b     = bi >> 10;
    const int lane  = threadIdx.x & 31;
    const int wave  = threadIdx.x >> 5;
    const int nloc  = lane & 15;               // N (output) lane within half
    const int khalf = lane >> 4;               // 0: K 0..7 / 1: K 8..15

    // B fragments: wf_w[16,64] as four 32x16 f16 tiles (upper K half = 0)
    v16h bfrag[4];
#pragma unroll
    for (int t = 0; t < 4; ++t) {
#pragma unroll
        for (int e = 0; e < 16; ++e)
            bfrag[t][e] = (e < 8) ? (_Float16)wf_w[(8 * khalf + e) * 64 + 16 * t + nloc]
                                  : (_Float16)0.f;
    }
    float biasv[4], afw[4];
#pragma unroll
    for (int t = 0; t < 4; ++t) {
        int o    = 16 * t + nloc;
        biasv[t] = wf_b[o] + bf[o];
        afw[t]   = af_w[o];
    }
    const float af_b = af_b_[0];
    const float e1i  = e1_[bi];
    const float* __restrict__ efrow = edge_feat + (size_t)bi * 1024 * 16;

    for (int it = 0; it < 8; ++it) {
        const int j0 = (wave * 8 + it) * 16;
        const float* src = efrow + (size_t)(j0 + nloc) * 16 + 8 * khalf;
        __builtin_prefetch(src + 16 * 16, 0, 1);      // next tile (global_prefetch_b8)

        // A fragment: 16 edges x K16 (f32 -> f16), zero-pad K 16..31
        const float4 x0 = *(const float4*)(src);
        const float4 x1 = *(const float4*)(src + 4);
        v16h a;
        a[0] = (_Float16)x0.x; a[1] = (_Float16)x0.y;
        a[2] = (_Float16)x0.z; a[3] = (_Float16)x0.w;
        a[4] = (_Float16)x1.x; a[5] = (_Float16)x1.y;
        a[6] = (_Float16)x1.z; a[7] = (_Float16)x1.w;
#pragma unroll
        for (int e = 8; e < 16; ++e) a[e] = (_Float16)0.f;

        v8f acc[4];
#pragma unroll
        for (int t = 0; t < 4; ++t) {
#pragma unroll
            for (int r = 0; r < 8; ++r) acc[t][r] = biasv[t];
            acc[t] = __builtin_amdgcn_wmma_f32_16x16x32_f16(
                false, a, false, bfrag[t], (short)0, acc[t], false, false);
        }

        // per-lane partial of  sum_o lrelu(f)*af_w[o]  (4 of 64 o's per lane)
        float partial[8];
#pragma unroll
        for (int r = 0; r < 8; ++r) {
            float p = lrelu(acc[0][r]) * afw[0];
            p += lrelu(acc[1][r]) * afw[1];
            p += lrelu(acc[2][r]) * afw[2];
            p += lrelu(acc[3][r]) * afw[3];
            partial[r] = p;
        }
        // butterfly o-reduction within each 16-lane half: all lanes get row sums
#pragma unroll
        for (int m = 1; m < 16; m <<= 1) {
#pragma unroll
            for (int r = 0; r < 8; ++r) partial[r] += __shfl_xor(partial[r], m, 32);
        }
        // 16 lanes (nloc<8 per half) each emit one row -> coalesced loads/stores
        if (nloc < 8) {
            float val = partial[0];
#pragma unroll
            for (int r = 1; r < 8; ++r) if (nloc == r) val = partial[r];
            const int   j  = j0 + 8 * khalf + nloc;
            const float cp = lrelu(val + af_b);
            const float cc = lrelu(e1i + e2_[b * 1024 + j]) + cp;
            const int   av = adj[(size_t)bi * 1024 + j];
            scores[(size_t)bi * 1024 + j] = (av > 0) ? cc : NEG_INF;
        }
    }
}

// ---------------------------------------------------------------------------
// Kernel 4 (WMMA): block = (b, 16-row i-tile). Softmax the 16 score rows
// (16-thread groups, shfl reductions, probs -> LDS f16), then
//   h_prime[16,64] = P[16,1024] @ h[1024,64]
// as WMMA: 8 waves split K=1024 (128 each), 4 k-chunks x 4 N-tiles of
// v_wmma_f32_16x16x32_f16; cross-wave combine via ds_add_f32 shared atomics.
// ---------------------------------------------------------------------------
__global__ __launch_bounds__(256)
void attn_kernel(const float* __restrict__ scores, const _Float16* __restrict__ hT16,
                 float* __restrict__ h_prime) {
    __shared__ __align__(16) _Float16 P[16 * 1024];   // probs, f16 (32 KB)
    __shared__ float hpacc[16 * 64];                  // cross-wave accumulator
    __shared__ float invden[16];

    const int blk = blockIdx.x;                // b*64 + itile
    const int b   = blk >> 6;
    const int i0  = (blk & 63) * 16;
    const int t   = threadIdx.x;
    const int lane = t & 31, wave = t >> 5;

    // ---- softmax: row = t/16 (16 threads per row, contiguous 64-elem chunks)
    const int row = t >> 4;                    // 0..15
    const int c   = t & 15;
    const float* __restrict__ sc = scores + (size_t)(b * 1024 + i0 + row) * 1024;

    float m = -INFINITY;
#pragma unroll
    for (int q = 0; q < 8; ++q) {
        const float4 x0 = *(const float4*)(sc + c * 64 + q * 8);
        const float4 x1 = *(const float4*)(sc + c * 64 + q * 8 + 4);
        m = fmaxf(m, fmaxf(fmaxf(x0.x, x0.y), fmaxf(x0.z, x0.w)));
        m = fmaxf(m, fmaxf(fmaxf(x1.x, x1.y), fmaxf(x1.z, x1.w)));
    }
#pragma unroll
    for (int s = 1; s < 16; s <<= 1) m = fmaxf(m, __shfl_xor(m, s, 32));

    float lsum = 0.f;
#pragma unroll
    for (int q = 0; q < 8; ++q) {
        const int k = c * 64 + q * 8;
        const float4 x0 = *(const float4*)(sc + k);
        const float4 x1 = *(const float4*)(sc + k + 4);
        float p0 = __expf(x0.x - m), p1 = __expf(x0.y - m);
        float p2 = __expf(x0.z - m), p3 = __expf(x0.w - m);
        float p4 = __expf(x1.x - m), p5 = __expf(x1.y - m);
        float p6 = __expf(x1.z - m), p7 = __expf(x1.w - m);
        v8h p;
        p[0] = (_Float16)p0; p[1] = (_Float16)p1; p[2] = (_Float16)p2; p[3] = (_Float16)p3;
        p[4] = (_Float16)p4; p[5] = (_Float16)p5; p[6] = (_Float16)p6; p[7] = (_Float16)p7;
        *(v8h*)&P[row * 1024 + k] = p;
        lsum += ((p0 + p1) + (p2 + p3)) + ((p4 + p5) + (p6 + p7));
    }
#pragma unroll
    for (int s = 1; s < 16; s <<= 1) lsum += __shfl_xor(lsum, s, 32);
    if (c == 0) invden[row] = 1.0f / lsum;

    for (int q = t; q < 1024; q += 256) hpacc[q] = 0.f;
    __syncthreads();

    // ---- WMMA GEMM: wave handles K in [wave*128, wave*128+128)
    const int nloc = lane & 15, khalf = lane >> 4;
    v8f acc[4];
#pragma unroll
    for (int tt = 0; tt < 4; ++tt)
#pragma unroll
        for (int r = 0; r < 8; ++r) acc[tt][r] = 0.f;

    const _Float16* __restrict__ hTb = hT16 + (size_t)b * 65536;
#pragma unroll
    for (int kc = 0; kc < 4; ++kc) {
        const int base = wave * 128 + kc * 32 + 8 * khalf;   // K base for this lane
        // A fragment from LDS probs: row M = nloc, K base..+7 and base+16..+23
        const v8h alo = *(const v8h*)&P[nloc * 1024 + base];
        const v8h ahi = *(const v8h*)&P[nloc * 1024 + base + 16];
        const v16h a = combine8(alo, ahi);
#pragma unroll
        for (int tt = 0; tt < 4; ++tt) {
            const _Float16* col = hTb + (size_t)(16 * tt + nloc) * 1024 + base;
            const v8h blo = *(const v8h*)(col);
            const v8h bhi = *(const v8h*)(col + 16);
            const v16h bb = combine8(blo, bhi);
            acc[tt] = __builtin_amdgcn_wmma_f32_16x16x32_f16(
                false, a, false, bb, (short)0, acc[tt], false, false);
        }
    }
    // cross-wave combine (ds_add_f32)
#pragma unroll
    for (int tt = 0; tt < 4; ++tt)
#pragma unroll
        for (int r = 0; r < 8; ++r) {
            const int rw = r + 8 * khalf, o = 16 * tt + nloc;
            atomicAdd(&hpacc[rw * 64 + o], acc[tt][r]);
        }
    __syncthreads();

    for (int q = t; q < 1024; q += 256) {
        const int rw = q >> 6, o = q & 63;
        h_prime[(size_t)(b * 1024 + i0 + rw) * 64 + o] = hpacc[q] * invden[rw];
    }
}

// ---------------------------------------------------------------------------
// Kernel 5: per node n, normalize over (b, o) axes (256 values, biased var)
// ---------------------------------------------------------------------------
__global__ __launch_bounds__(256)
void norm_kernel(const float* __restrict__ h_prime, float* __restrict__ out) {
    __shared__ float s1[256], s2[256];
    const int n = blockIdx.x, t = threadIdx.x;
    const int b = t >> 6, o = t & 63;
    const float x = h_prime[((size_t)b * 1024 + n) * 64 + o];
    s1[t] = x; s2[t] = x * x; __syncthreads();
    for (int s = 128; s > 0; s >>= 1) {
        if (t < s) { s1[t] += s1[t + s]; s2[t] += s2[t + s]; }
        __syncthreads();
    }
    const float mean = s1[0] * (1.0f / 256.0f);
    const float var  = s2[0] * (1.0f / 256.0f) - mean * mean;
    out[((size_t)b * 1024 + n) * 64 + o] = (x - mean) * rsqrtf(1e-5f + var);
}

// ---------------------------------------------------------------------------
extern "C" void kernel_launch(void* const* d_in, const int* in_sizes, int n_in,
                              void* d_out, int out_size, void* d_ws, size_t ws_size,
                              hipStream_t stream) {
    const float* input     = (const float*)d_in[0];   // [4,1024,128]
    const float* edge_feat = (const float*)d_in[1];   // [4,1024,1024,16]
    const int*   adj       = (const int*)  d_in[2];   // [4,1024,1024]
    const float* wh_w      = (const float*)d_in[3];
    const float* wh_b      = (const float*)d_in[4];
    const float* wh1_w     = (const float*)d_in[5];
    const float* wh1_b     = (const float*)d_in[6];
    const float* wh2_w     = (const float*)d_in[7];
    const float* wh2_b     = (const float*)d_in[8];
    const float* ah_w      = (const float*)d_in[9];
    const float* ah_b      = (const float*)d_in[10];
    const float* wf_w      = (const float*)d_in[11];
    const float* wf_b      = (const float*)d_in[12];
    const float* af_w      = (const float*)d_in[13];
    const float* af_b      = (const float*)d_in[14];
    const float* bf        = (const float*)d_in[15];

    char* ws = (char*)d_ws;
    float*     scores  = (float*)    (ws);                          // 16 MB
    _Float16*  hT16    = (_Float16*) (ws + 16777216);               // 512 KB, [4][64][1024]
    float*     h_prime = (float*)    (ws + 16777216 + 524288);      // 1 MB
    float*     e1      = (float*)    (ws + 16777216 + 524288 + 1048576);
    float*     e2      = (float*)    (ws + 16777216 + 524288 + 1048576 + 16384);
    float*     v1      = (float*)    (ws + 16777216 + 524288 + 1048576 + 32768);
    float*     v2      = (float*)    (ws + 16777216 + 524288 + 1048576 + 32768 + 512);
    float*     b12     = (float*)    (ws + 16777216 + 524288 + 1048576 + 32768 + 1024);

    fold_kernel<<<1, 128, 0, stream>>>(wh1_w, wh1_b, wh2_w, wh2_b, ah_w, ah_b, v1, v2, b12);
    proj_kernel<<<4096, 128, 0, stream>>>(input, wh_w, wh_b, v1, v2, b12, hT16, e1, e2);
    score_kernel<<<4096, 256, 0, stream>>>(edge_feat, adj, wf_w, wf_b, af_w, af_b, bf,
                                           e1, e2, scores);
    attn_kernel<<<256, 256, 0, stream>>>(scores, hT16, h_prime);
    norm_kernel<<<1024, 256, 0, stream>>>(h_prime, (float*)d_out);
}
